// RoPE_Attn_15144054686376
// MI455X (gfx1250) — matrix operand order repaired and checked by reading, not compiled, not run on hardware
//
#include <hip/hip_runtime.h>
#include <hip/hip_bf16.h>

// ---------------------------------------------------------------------------
// Problem constants (B, S, H, E) = (2, 2048, 16, 2048), D = 128
// ---------------------------------------------------------------------------
#define BB 2
#define SS 2048
#define HH 16
#define EE 2048
#define DD 128
#define FF (3 * EE)   // 6144 qkv rows of Wc
#define MM (BB * SS)  // 4096 token rows

typedef __attribute__((ext_vector_type(16))) __bf16 v16bf;
typedef __attribute__((ext_vector_type(8)))  float  v8f;
typedef __attribute__((ext_vector_type(4)))  int    si4;
typedef __attribute__((ext_vector_type(8)))  int    si8;

static __device__ __forceinline__ unsigned short f2bf(float f) {
  unsigned u = __builtin_bit_cast(unsigned, f);
  u += 0x7FFFu + ((u >> 16) & 1u);          // round-to-nearest-even
  return (unsigned short)(u >> 16);
}

union Frag {
  v16bf v;
  uint4 q[2];
};

static __device__ __forceinline__ v8f wmma_bf16(const Frag &a, const Frag &b, v8f c) {
  // D = A(16x32 bf16) * B(32x16 bf16) + C(16x16 f32)
  return __builtin_amdgcn_wmma_f32_16x16x32_bf16(false, a.v, false, b.v,
                                                 (short)0, c, false, false);
}

// ---------------------------------------------------------------------------
// Tensor Data Mover: issue a 2D bf16 tile load (global -> LDS).
// D# built per CDNA5 ISA §8: group0 = {count|flags, lds_addr, gaddr lo,
// gaddr hi | type=2}; group1 = {data_size, tensor dims, tile dims, stride}.
// Wave-uniform SGPR descriptors, VIMAGE 4-group form (groups 2/3 zeroed).
// ---------------------------------------------------------------------------
static __device__ __forceinline__ void tdm_load_2d_bf16(
    unsigned lds_byte_off, unsigned long long gaddr,
    unsigned tile_d0, unsigned tile_d1, unsigned stride0,
    unsigned tensor_d0, unsigned tensor_d1)
{
  si4 g0;
  g0[0] = 1;                                            // count=1, user mode
  g0[1] = (int)lds_byte_off;                            // LDS byte address
  g0[2] = (int)(unsigned)(gaddr & 0xffffffffull);       // global addr [31:0]
  g0[3] = (int)(((unsigned)(gaddr >> 32)) & 0x1ffffffu) // global addr [56:32]
        | (int)0x80000000;                              // type = 2 ("image")
  si8 g1;
  g1[0] = 1 << 16;                                      // data_size=1 (2 bytes)
  g1[1] = (int)((tensor_d0 & 0xffffu) << 16);           // tensor_dim0 [15:0]
  g1[2] = (int)((tensor_d0 >> 16) | ((tensor_d1 & 0xffffu) << 16));
  g1[3] = (int)((tensor_d1 >> 16) | (tile_d0 << 16));   // tile_dim0
  g1[4] = (int)tile_d1;                                 // tile_dim1 (tile_dim2=0)
  g1[5] = (int)stride0;                                 // tensor_dim0_stride lo
  g1[6] = 0;                                            // stride hi / dim1_stride
  g1[7] = 0;
  si4 gz = {0, 0, 0, 0};
  asm volatile("tensor_load_to_lds %0, %1, %2, %3"
               :: "s"(g0), "s"(g1), "s"(gz), "s"(gz)
               : "memory");
}

// ---------------------------------------------------------------------------
// Kernel 1: qkv = x @ Wc^T + bc, fused rope-mult + scatter.
// Block tile 128(M) x 128(N), K-step 64. 8 waves stacked along M: each wave
// computes 16 rows x 128 cols = one full head slice -> rope pairing (d, d-64)
// is tile (t, t-4) in the same lane.
// ---------------------------------------------------------------------------
#define AS_STRIDE 72    // 64 + 8 pad (ushorts), 144B row = 16B multiple
#define BS_STRIDE 136   // 128 + 8 pad (ushorts), 272B row = 16B multiple

__global__ __launch_bounds__(256) void qkv_gemm_rope_kernel(
    const float* __restrict__ x, const float* __restrict__ Wc,
    const float* __restrict__ bc, const float* __restrict__ freqs,
    float* __restrict__ out_k, float* __restrict__ out_v,
    unsigned short* __restrict__ q_bf,   // (B,H,S,D) bf16, pre-scaled 1/sqrt(D)
    unsigned short* __restrict__ kT_bf,  // (B,H,D,S) bf16 (roped)
    unsigned short* __restrict__ v_bf)   // (B,H,S,D) bf16
{
  __shared__ unsigned short As[128 * AS_STRIDE];  // [m][k] bf16
  __shared__ unsigned short Bs[64 * BS_STRIDE];   // [k][n] bf16 (transposed Wc)

  const int tid  = threadIdx.x;
  const int lane = tid & 31;
  const int wave = tid >> 5;
  const int col  = lane & 15;
  const int hi   = lane >> 4;
  const int blockM = blockIdx.x * 128;
  const int blockN = blockIdx.y * 128;

  const int lrow = tid >> 1;        // 0..127: tile row loaded by this thread
  const int lks  = (tid & 1) * 32;  // K half loaded by this thread

  v8f acc[8];
  #pragma unroll
  for (int t = 0; t < 8; ++t) acc[t] = (v8f)0.0f;

  const float* xrow = x  + (size_t)(blockM + lrow) * EE;
  const float* wrow = Wc + (size_t)(blockN + lrow) * EE;

  for (int k0 = 0; k0 < EE; k0 += 64) {
    float4 av[8], bv[8];
    #pragma unroll
    for (int i = 0; i < 8; ++i) {
      av[i] = *(const float4*)(xrow + k0 + lks + 4 * i);
      bv[i] = *(const float4*)(wrow + k0 + lks + 4 * i);
    }
    if (k0 + 64 < EE) {
      __builtin_prefetch(xrow + k0 + 64 + lks, 0, 0);  // global_prefetch_b8
      __builtin_prefetch(wrow + k0 + 64 + lks, 0, 0);
    }
    __syncthreads();   // previous tile fully consumed
    #pragma unroll
    for (int i = 0; i < 8; ++i) {
      const int kk = lks + 4 * i;
      unsigned p0 = (unsigned)f2bf(av[i].x) | ((unsigned)f2bf(av[i].y) << 16);
      unsigned p1 = (unsigned)f2bf(av[i].z) | ((unsigned)f2bf(av[i].w) << 16);
      *(unsigned*)(&As[lrow * AS_STRIDE + kk])     = p0;
      *(unsigned*)(&As[lrow * AS_STRIDE + kk + 2]) = p1;
      Bs[(kk + 0) * BS_STRIDE + lrow] = f2bf(bv[i].x);  // transpose into [k][n]
      Bs[(kk + 1) * BS_STRIDE + lrow] = f2bf(bv[i].y);
      Bs[(kk + 2) * BS_STRIDE + lrow] = f2bf(bv[i].z);
      Bs[(kk + 3) * BS_STRIDE + lrow] = f2bf(bv[i].w);
    }
    __syncthreads();
    #pragma unroll
    for (int ks = 0; ks < 2; ++ks) {
      // A frag: lane&15 = M row; K chunks {kb..kb+7, kb+16..kb+23}, kb = hi*8
      Frag a;
      const int arow = wave * 16 + col;
      const int kb   = ks * 32 + hi * 8;
      a.q[0] = *(const uint4*)(&As[arow * AS_STRIDE + kb]);
      a.q[1] = *(const uint4*)(&As[arow * AS_STRIDE + kb + 16]);
      // Preload B frags in groups of 4 so one dscnt wait covers 4 WMMAs
      #pragma unroll
      for (int g = 0; g < 2; ++g) {
        Frag bf[4];
        #pragma unroll
        for (int t = 0; t < 4; ++t) {
          const int n0 = (g * 4 + t) * 16;
          #pragma unroll
          for (int s = 0; s < 16; ++s) {
            const int brow = ks * 32 + hi * 8 + (s & 7) + (s >> 3) * 16;
            bf[t].v[s] = __builtin_bit_cast(__bf16, Bs[brow * BS_STRIDE + n0 + col]);
          }
        }
        #pragma unroll
        for (int t = 0; t < 4; ++t)
          acc[g * 4 + t] = wmma_bf16(a, bf[t], acc[g * 4 + t]);
      }
    }
  }

  // ---- Epilogue: bias + rope-mult + scatter --------------------------------
  const int sec = blockN >> 11;            // 0=q, 1=k, 2=v
  const int h   = (blockN & 2047) >> 7;    // head (blockN is 128-aligned)
  const int waveM = blockM + wave * 16;
  const float invsq = 0.08838834764831845f;  // 1/sqrt(128)

  float bias[8];
  #pragma unroll
  for (int t = 0; t < 8; ++t) bias[t] = bc[blockN + t * 16 + col];

  #pragma unroll
  for (int r = 0; r < 8; ++r) {
    const int m = waveM + r + hi * 8;      // C layout: VGPR r -> rows r / r+8
    const int b = m >> 11;
    const int s = m & 2047;
    const size_t bhs = (size_t)(b * HH + h) * SS + s;
    float vr[8];
    #pragma unroll
    for (int t = 0; t < 8; ++t) vr[t] = acc[t][r] + bias[t];
    if (sec == 2) {
      #pragma unroll
      for (int t = 0; t < 8; ++t) {
        const int d = t * 16 + col;
        out_v[bhs * DD + d] = vr[t];
        v_bf[bhs * DD + d]  = f2bf(vr[t]);
      }
    } else {
      float fr[4];
      #pragma unroll
      for (int t = 0; t < 4; ++t) fr[t] = freqs[s * 64 + t * 16 + col];
      #pragma unroll
      for (int t = 0; t < 8; ++t) {
        const int d = t * 16 + col;
        const float rv = vr[t & 3] * fr[t & 3];  // concat([xh, xh]) * f
        if (sec == 0) {
          q_bf[bhs * DD + d] = f2bf(rv * invsq); // fold softmax scale into Q
        } else {
          out_k[bhs * DD + d] = rv;
          kT_bf[((size_t)(b * HH + h) * DD + d) * SS + s] = f2bf(rv);
        }
      }
    }
  }
}

// ---------------------------------------------------------------------------
// Kernel 2: flash attention. 1 wave = 16 queries x full D=128, online softmax
// over 32-key tiles. K/V tiles are block-shared: wave 0 DMAs them into LDS
// with the Tensor Data Mover (double-buffered), everyone syncs on a barrier.
//
// LDS arena (ushorts):
//   [0      .. 8191 ]  K tiles  2 x [128 d rows][32 keys]   (8 KB each)
//   [8192   .. 16383]  V tiles  2 x [32 key rows][128 d]    (8 KB each)
//   [16384  .. 20479]  per-wave P re-layout buffers 8 x 16x32
// ---------------------------------------------------------------------------
__global__ __launch_bounds__(256) void attn_kernel(
    const unsigned short* __restrict__ q_bf,
    const unsigned short* __restrict__ kT_bf,
    const unsigned short* __restrict__ v_bf,
    float* __restrict__ y)
{
  __shared__ unsigned short smem[2 * 128 * 32 + 2 * 32 * 128 + 8 * 16 * 32];

  const int tid  = threadIdx.x;
  const int lane = tid & 31;
  const int wave = tid >> 5;
  const int col  = lane & 15;
  const int hi   = lane >> 4;

  const int bh   = blockIdx.x >> 4;   // 0..31 (b*H + h)
  const int qblk = blockIdx.x & 15;
  const int q0   = qblk * 128 + wave * 16;
  const int b    = bh >> 4;
  const int h    = bh & 15;

  const unsigned short* kT_base = kT_bf + (size_t)bh * DD * SS;  // (D,S)
  const unsigned short* v_base  = v_bf  + (size_t)bh * SS * DD;  // (S,D)

  // Persistent Q A-fragments (4 K-steps covering D=128)
  Frag qf[4];
  {
    const unsigned short* qp = q_bf + ((size_t)bh * SS + (q0 + col)) * DD;
    #pragma unroll
    for (int ks = 0; ks < 4; ++ks) {
      const int kb = ks * 32 + hi * 8;
      qf[ks].q[0] = *(const uint4*)(qp + kb);
      qf[ks].q[1] = *(const uint4*)(qp + kb + 16);
    }
  }

  float mrow[8], lrow[8];
  v8f acc[8];
  #pragma unroll
  for (int r = 0; r < 8; ++r) { mrow[r] = -3.0e38f; lrow[r] = 0.0f; }
  #pragma unroll
  for (int t = 0; t < 8; ++t) acc[t] = (v8f)0.0f;

  unsigned short* pb = smem + 16384 + wave * (16 * 32);

  // Prologue: DMA first K/V tiles into buffer 0 (TENSORcnt-tracked).
  if (wave == 0) {
    tdm_load_2d_bf16(/*lds*/ 0u,     (unsigned long long)kT_base,
                     /*tile*/ 32, 128, /*stride*/ SS, SS, DD);
    tdm_load_2d_bf16(/*lds*/ 16384u, (unsigned long long)v_base,
                     /*tile*/ 128, 32, /*stride*/ DD, DD, SS);
  }

  for (int it = 0; it < SS / 32; ++it) {
    const int buf = it & 1;
    const int kt  = it * 32;
    if (wave == 0) {
      if (it + 1 < SS / 32) {
        const int kt2 = kt + 32;
        const int nb  = buf ^ 1;
        tdm_load_2d_bf16((unsigned)(nb * 8192),
                         (unsigned long long)(kT_base + kt2),
                         32, 128, SS, SS, DD);
        tdm_load_2d_bf16((unsigned)(16384 + nb * 8192),
                         (unsigned long long)(v_base + (size_t)kt2 * DD),
                         128, 32, DD, DD, SS);
        __builtin_amdgcn_s_wait_tensorcnt(2);  // current pair complete
      } else {
        __builtin_amdgcn_s_wait_tensorcnt(0);
      }
    }
    __syncthreads();   // current buffer published to all waves

    const unsigned short* Kls = smem + buf * 4096;          // [128][32]
    const unsigned short* Vls = smem + 8192 + buf * 4096;   // [32][128]

    // ---- scores S = Q K^T (16 x 32) -------------------------------------
    v8f s0 = (v8f)0.0f, s1 = (v8f)0.0f;
    #pragma unroll
    for (int ks = 0; ks < 4; ++ks) {
      Frag b0, b1;
      #pragma unroll
      for (int s = 0; s < 16; ++s) {
        const int d = ks * 32 + hi * 8 + (s & 7) + (s >> 3) * 16;
        b0.v[s] = __builtin_bit_cast(__bf16, Kls[d * 32 + col]);
        b1.v[s] = __builtin_bit_cast(__bf16, Kls[d * 32 + 16 + col]);
      }
      s0 = wmma_bf16(qf[ks], b0, s0);
      s1 = wmma_bf16(qf[ks], b1, s1);
    }
    // ---- online softmax (row ops across the 16-lane group) --------------
    float tm[8], p0[8], p1[8], sc[8], rs[8];
    #pragma unroll
    for (int r = 0; r < 8; ++r) tm[r] = fmaxf(s0[r], s1[r]);
    #pragma unroll
    for (int off = 8; off >= 1; off >>= 1) {
      #pragma unroll
      for (int r = 0; r < 8; ++r) tm[r] = fmaxf(tm[r], __shfl_xor(tm[r], off, 32));
    }
    #pragma unroll
    for (int r = 0; r < 8; ++r) {
      const float mn = fmaxf(mrow[r], tm[r]);
      sc[r] = __expf(mrow[r] - mn);
      mrow[r] = mn;
      p0[r] = __expf(s0[r] - mn);
      p1[r] = __expf(s1[r] - mn);
      rs[r] = p0[r] + p1[r];
    }
    #pragma unroll
    for (int off = 8; off >= 1; off >>= 1) {
      #pragma unroll
      for (int r = 0; r < 8; ++r) rs[r] += __shfl_xor(rs[r], off, 32);
    }
    #pragma unroll
    for (int r = 0; r < 8; ++r) lrow[r] = lrow[r] * sc[r] + rs[r];
    #pragma unroll
    for (int t = 0; t < 8; ++t) {
      #pragma unroll
      for (int r = 0; r < 8; ++r) acc[t][r] *= sc[r];
    }
    // ---- C-layout P -> LDS -> A-layout fragment -------------------------
    #pragma unroll
    for (int r = 0; r < 8; ++r) {
      const int row = r + hi * 8;
      pb[row * 32 + col]      = f2bf(p0[r]);
      pb[row * 32 + 16 + col] = f2bf(p1[r]);
    }
    Frag pf;
    {
      const int kb = hi * 8;
      pf.q[0] = *(const uint4*)(&pb[col * 32 + kb]);
      pf.q[1] = *(const uint4*)(&pb[col * 32 + kb + 16]);
    }
    // ---- acc += P @ V (V from LDS), pairs share one dscnt wait ----------
    #pragma unroll
    for (int g = 0; g < 4; ++g) {
      Frag vb[2];
      #pragma unroll
      for (int t = 0; t < 2; ++t) {
        #pragma unroll
        for (int s = 0; s < 16; ++s) {
          const int key = hi * 8 + (s & 7) + (s >> 3) * 16;
          vb[t].v[s] = __builtin_bit_cast(__bf16, Vls[key * 128 + (g * 2 + t) * 16 + col]);
        }
      }
      #pragma unroll
      for (int t = 0; t < 2; ++t)
        acc[g * 2 + t] = wmma_bf16(pf, vb[t], acc[g * 2 + t]);
    }
    __syncthreads();   // all waves done with this buffer before next DMA lands
  }

  // ---- normalize and write y (B*S, E) ------------------------------------
  #pragma unroll
  for (int r = 0; r < 8; ++r) {
    const int sq = q0 + r + hi * 8;
    const float inv = 1.0f / lrow[r];
    float* yr = y + (size_t)(b * SS + sq) * EE + h * DD;
    #pragma unroll
    for (int t = 0; t < 8; ++t) yr[t * 16 + col] = acc[t][r] * inv;
  }
}

// ---------------------------------------------------------------------------
extern "C" void kernel_launch(void* const* d_in, const int* in_sizes, int n_in,
                              void* d_out, int out_size, void* d_ws, size_t ws_size,
                              hipStream_t stream) {
  const float* x     = (const float*)d_in[0];
  const float* Wc    = (const float*)d_in[1];
  const float* bc    = (const float*)d_in[2];
  const float* freqs = (const float*)d_in[3];

  float* y     = (float*)d_out;                       // (B*S, E)
  float* out_k = y + (size_t)MM * EE;                 // (B,H,S,D)
  float* out_v = out_k + (size_t)BB * HH * SS * DD;   // (B,H,S,D)

  unsigned short* q_bf  = (unsigned short*)d_ws;                 // 16 MB
  unsigned short* kT_bf = q_bf + (size_t)BB * HH * SS * DD;      // 16 MB
  unsigned short* v_bf  = kT_bf + (size_t)BB * HH * SS * DD;     // 16 MB

  dim3 g1(MM / 128, FF / 128);   // 32 x 48
  qkv_gemm_rope_kernel<<<g1, dim3(256), 0, stream>>>(
      x, Wc, bc, freqs, out_k, out_v, q_bf, kT_bf, v_bf);

  dim3 g2(BB * HH * (SS / 128)); // 512
  attn_kernel<<<g2, dim3(256), 0, stream>>>(q_bf, kT_bf, v_bf, y);
}
